// GRU_40381282517738
// MI455X (gfx1250) — compile-verified, matrix operand order checked by reading, not run on hardware
//
#include <hip/hip_runtime.h>
#include <hip/hip_bf16.h>

// ---------------------------------------------------------------------------
// GRU for MI455X (gfx1250): bf16 WMMA GEMMs + fused per-step scan kernels.
//   B=64, S=512, NI=NH=1024
//   - Phase 1: 128x128-tile bf16 WMMA GEMM for the input projections.
//   - Phase 2: 512 sequential steps; step_rz (manual LDS staging) and
//     step_h (Tensor Data Mover staging of the weight tile).
// ---------------------------------------------------------------------------

typedef __bf16 bf16_t;
typedef bf16_t bf16x16 __attribute__((ext_vector_type(16)));
typedef float  f32x8  __attribute__((ext_vector_type(8)));
typedef unsigned int u32x4 __attribute__((ext_vector_type(4)));
typedef int i32x4v __attribute__((ext_vector_type(4)));
typedef int i32x8v __attribute__((ext_vector_type(8)));

#define B_  64
#define S_  512
#define NI_ 1024
#define NH_ 1024

__device__ __forceinline__ bf16_t to_bf16(float f) {
  union { float f; unsigned u; } v; v.f = f;
  unsigned r = (v.u + 0x7FFFu + ((v.u >> 16) & 1u)) >> 16;
  unsigned short s = (unsigned short)r;
  return __builtin_bit_cast(bf16_t, s);
}

__device__ __forceinline__ float sigmoidf_(float x) {
  return 1.0f / (1.0f + __expf(-x));
}

// Load a 16x32 bf16 operand fragment from a K-contiguous LDS tile.
// CDNA5 16-bit A/B layout: lanes 0-15 -> rows 0-15, K {0..7,16..23};
//                          lanes 16-31 -> rows 0-15, K {8..15,24..31}.
__device__ __forceinline__ bf16x16 load_frag(const bf16_t* base, int ldk) {
  const int lane = threadIdx.x & 31;
  const bf16_t* p = base + (lane & 15) * ldk + ((lane >> 4) << 3);
  bf16x16 f;
#pragma unroll
  for (int i = 0; i < 8; ++i) f[i] = p[i];          // K half 0
#pragma unroll
  for (int i = 0; i < 8; ++i) f[8 + i] = p[16 + i]; // K half 1
  return f;
}

// Same fragment but gathered from a row-major [K][N] LDS tile (TDM output;
// TDM cannot transpose, so we transpose on read).
__device__ __forceinline__ bf16x16 load_frag_rowmajor(const bf16_t* base, int ldn) {
  const int lane = threadIdx.x & 31;
  const int col = lane & 15;
  const int kh  = (lane >> 4) << 3;  // 0 or 8
  bf16x16 f;
#pragma unroll
  for (int i = 0; i < 8; ++i) f[i]     = base[(kh + i) * ldn + col];
#pragma unroll
  for (int i = 0; i < 8; ++i) f[8 + i] = base[(kh + 16 + i) * ldn + col];
  return f;
}

__device__ __forceinline__ f32x8 wmma_bf16(bf16x16 a, bf16x16 b, f32x8 c) {
  return __builtin_amdgcn_wmma_f32_16x16x32_bf16(false, a, false, b,
                                                 (short)0, c, false, false);
}

__device__ __forceinline__ f32x8 zero8() {
  f32x8 z = {0.f, 0.f, 0.f, 0.f, 0.f, 0.f, 0.f, 0.f};
  return z;
}

// ---------------------------------------------------------------------------
// Tensor Data Mover: DMA a 2D bf16 tile (tile_d1 rows x tile_d0 cols) from a
// row-major global tensor (row length tensor_d0, row stride d0_stride elems)
// into LDS at lds_addr. D# packing per CDNA5 ISA ch.8 (groups 0/1; 2/3 unused).
// amdgpu-toolchain (clang-23) 6-arg builtin form.
// ---------------------------------------------------------------------------
__device__ __forceinline__ void tdm_load_2d_bf16(
    unsigned lds_addr, const void* gaddr,
    unsigned tensor_d0, unsigned tensor_d1, unsigned d0_stride,
    unsigned tile_d0, unsigned tile_d1) {
  unsigned long long ga = (unsigned long long)gaddr;
  u32x4 g0;
  g0[0] = 1u;                                          // count=1, user mode
  g0[1] = lds_addr;                                    // LDS byte address
  g0[2] = (unsigned)(ga & 0xFFFFFFFFu);                // global addr lo
  g0[3] = (unsigned)((ga >> 32) & 0x1FFFFFFu)          // global addr hi (57b)
        | (2u << 30);                                  // type = 2 (image)
  i32x8v g1;
  g1[0] = (int)(1u << 16);                             // data_size=1 (2 bytes)
  g1[1] = (int)((tensor_d0 & 0xFFFFu) << 16);          // tensor_dim0[15:0]
  g1[2] = (int)(((tensor_d0 >> 16) & 0xFFFFu)          // tensor_dim0[31:16]
        | ((tensor_d1 & 0xFFFFu) << 16));              // tensor_dim1[15:0]
  g1[3] = (int)(((tensor_d1 >> 16) & 0xFFFFu)          // tensor_dim1[31:16]
        | ((tile_d0 & 0xFFFFu) << 16));                // tile_dim0
  g1[4] = (int)(tile_d1 & 0xFFFFu);                    // tile_dim1 (tile_dim2=0)
  g1[5] = (int)d0_stride;                              // tensor_dim0_stride lo32
  g1[6] = 0;                                           // stride hi / dim1_stride
  g1[7] = 0;
  i32x4v gz4 = {0, 0, 0, 0};
  i32x8v gz8 = {0, 0, 0, 0, 0, 0, 0, 0};
  __builtin_amdgcn_tensor_load_to_lds(g0, g1, gz4, gz4, gz8, 0);
}

// ---------------------------------------------------------------------------
// prep: fp32 weights -> bf16 (packed [Wxz,Wxr,Wxh,Whz,Whr,Whh]); copy H0.
// ---------------------------------------------------------------------------
__global__ void prep_kernel(const float* __restrict__ Wxz, const float* __restrict__ Whz,
                            const float* __restrict__ Wxr, const float* __restrict__ Whr,
                            const float* __restrict__ Wxh, const float* __restrict__ Whh,
                            const float* __restrict__ Hin,
                            bf16_t* __restrict__ wb, float* __restrict__ Hf,
                            bf16_t* __restrict__ Hb) {
  const long WN = (long)NI_ * NH_;
  long i = (long)blockIdx.x * blockDim.x + threadIdx.x;
  if (i < 6 * WN) {
    int which = (int)(i / WN);
    long off = i - (long)which * WN;
    const float* src = (which == 0) ? Wxz : (which == 1) ? Wxr : (which == 2) ? Wxh
                     : (which == 3) ? Whz : (which == 4) ? Whr : Whh;
    wb[i] = to_bf16(src[off]);
  } else {
    long j = i - 6 * WN;
    if (j < (long)B_ * NH_) {
      float h = Hin[j];
      Hf[j] = h;
      Hb[j] = to_bf16(h);
    }
  }
}

// ---------------------------------------------------------------------------
// Phase 1: X(32768x1024) @ W_x{z,r,h}(1024x1024) + b.
// 128x128 tile / workgroup, 8 waves, each wave 32x64 (2x4 wmma accums).
// gate 0 -> xz(bf16 ws), 1 -> xr(bf16 ws), 2 -> xh written into d_out (fp32).
// ---------------------------------------------------------------------------
__global__ __launch_bounds__(256) void gemm_x_kernel(
    const float* __restrict__ X, const bf16_t* __restrict__ wb,
    const float* __restrict__ bz, const float* __restrict__ br,
    const float* __restrict__ bh,
    bf16_t* __restrict__ xz, bf16_t* __restrict__ xr, float* __restrict__ xh) {
  constexpr int LDK = 40;                       // 32 + pad, keeps 16B align
  __shared__ __attribute__((aligned(16))) bf16_t sA[128 * LDK];
  __shared__ __attribute__((aligned(16))) bf16_t sBt[128 * LDK];   // transposed [n][k]

  const int gate = blockIdx.z;
  const int rowBase = blockIdx.x * 128;
  const int colBase = blockIdx.y * 128;
  const bf16_t* W = wb + (long)gate * NI_ * NH_;              // [K][N] row-major
  const float* bias = (gate == 0) ? bz : (gate == 1) ? br : bh;

  const int tid = threadIdx.x;
  const int lane = tid & 31;
  const int wave = tid >> 5;
  const int wr = wave >> 1;   // 0..3  -> M offset wr*32
  const int wc = wave & 1;    // 0..1  -> N offset wc*64

  f32x8 acc[2][4];
#pragma unroll
  for (int i = 0; i < 2; ++i)
#pragma unroll
    for (int j = 0; j < 4; ++j) acc[i][j] = zero8();

  const int arow = tid >> 1;        // 0..127
  const int akp  = (tid & 1) * 16;  // 0 / 16
  const int bk   = tid >> 3;        // 0..31
  const int bnp  = (tid & 7) * 16;  // 0..112

  for (int k0 = 0; k0 < NI_; k0 += 32) {
    // ---- stage A tile: fp32 -> bf16 ----
    {
      const float4* src = (const float4*)(X + (long)(rowBase + arow) * NI_ + k0 + akp);
      if (k0 + 32 < NI_)
        __builtin_prefetch(X + (long)(rowBase + arow) * NI_ + k0 + 32 + akp, 0, 1);
#pragma unroll
      for (int v = 0; v < 4; ++v) {
        float4 d = src[v];
        bf16_t* dst = &sA[arow * LDK + akp + v * 4];
        dst[0] = to_bf16(d.x); dst[1] = to_bf16(d.y);
        dst[2] = to_bf16(d.z); dst[3] = to_bf16(d.w);
      }
    }
    // ---- stage B tile transposed: sBt[n][k] ----
    {
      const uint4* src = (const uint4*)(W + (long)(k0 + bk) * NH_ + colBase + bnp);
      uint4 d0 = src[0], d1 = src[1];
      const bf16_t* e0 = (const bf16_t*)&d0;
      const bf16_t* e1 = (const bf16_t*)&d1;
#pragma unroll
      for (int j = 0; j < 8; ++j) {
        sBt[(bnp + j)     * LDK + bk] = e0[j];
        sBt[(bnp + 8 + j) * LDK + bk] = e1[j];
      }
    }
    __syncthreads();

    bf16x16 aF[2], bF[4];
#pragma unroll
    for (int i = 0; i < 2; ++i) aF[i] = load_frag(&sA[(wr * 32 + i * 16) * LDK], LDK);
#pragma unroll
    for (int j = 0; j < 4; ++j) bF[j] = load_frag(&sBt[(wc * 64 + j * 16) * LDK], LDK);
#pragma unroll
    for (int i = 0; i < 2; ++i)
#pragma unroll
      for (int j = 0; j < 4; ++j) acc[i][j] = wmma_bf16(aF[i], bF[j], acc[i][j]);
    __syncthreads();
  }

  // epilogue: C/D layout VGPR e: lanes 0-15 -> M=e, lanes 16-31 -> M=e+8.
  // Gate branch hoisted outside the store loops (uniform per block).
  const int mlane = (lane >> 4) * 8;
  const int nlane = lane & 15;
  if (gate == 2) {
#pragma unroll
    for (int i = 0; i < 2; ++i)
#pragma unroll
      for (int j = 0; j < 4; ++j) {
        int n = colBase + wc * 64 + j * 16 + nlane;
        float bv = bias[n];
#pragma unroll
        for (int e = 0; e < 8; ++e) {
          int m = rowBase + wr * 32 + i * 16 + mlane + e;
          xh[(long)m * NH_ + n] = acc[i][j][e] + bv;   // into d_out (B,S,NH)
        }
      }
  } else {
    bf16_t* dst = (gate == 0) ? xz : xr;
#pragma unroll
    for (int i = 0; i < 2; ++i)
#pragma unroll
      for (int j = 0; j < 4; ++j) {
        int n = colBase + wc * 64 + j * 16 + nlane;
        float bv = bias[n];
#pragma unroll
        for (int e = 0; e < 8; ++e) {
          int m = rowBase + wr * 32 + i * 16 + mlane + e;
          dst[(long)m * NH_ + n] = to_bf16(acc[i][j][e] + bv);
        }
      }
  }
}

// ---------------------------------------------------------------------------
// Step stage 1: r = sigmoid(xr_t + H@W_hr), z = sigmoid(xz_t + H@W_hz),
//               RH = bf16(r * H), Z = z.   M=64, N=1024, K=1024.
// One workgroup per 128 cols; 8 waves; wave = 16 rows x 64 cols; 2 gates fused.
// ---------------------------------------------------------------------------
__global__ __launch_bounds__(256) void step_rz_kernel(
    const bf16_t* __restrict__ Hb, const float* __restrict__ Hf,
    const bf16_t* __restrict__ Whr, const bf16_t* __restrict__ Whz,
    const bf16_t* __restrict__ xr, const bf16_t* __restrict__ xz,
    bf16_t* __restrict__ RH, float* __restrict__ Zbuf, int t) {
  constexpr int LDK = 40;
  __shared__ __attribute__((aligned(16))) bf16_t sA[64 * LDK];
  __shared__ __attribute__((aligned(16))) bf16_t sBr[128 * LDK];
  __shared__ __attribute__((aligned(16))) bf16_t sBz[128 * LDK];

  const int colBase = blockIdx.x * 128;
  const int tid = threadIdx.x;
  const int lane = tid & 31;
  const int wave = tid >> 5;
  const int wr = wave >> 1;   // 0..3 -> 16-row slice of M=64
  const int wc = wave & 1;

  f32x8 accR[4], accZ[4];
#pragma unroll
  for (int j = 0; j < 4; ++j) { accR[j] = zero8(); accZ[j] = zero8(); }

  const int arow = tid >> 2;        // 0..63
  const int akp  = (tid & 3) * 8;   // 0..24
  const int bk   = tid >> 3;        // 0..31
  const int bnp  = (tid & 7) * 16;  // 0..112

  for (int k0 = 0; k0 < NH_; k0 += 32) {
    {
      uint4 d = *(const uint4*)(Hb + (long)arow * NH_ + k0 + akp);
      const bf16_t* e = (const bf16_t*)&d;
#pragma unroll
      for (int j = 0; j < 8; ++j) sA[arow * LDK + akp + j] = e[j];
    }
    {
      const uint4* srcR = (const uint4*)(Whr + (long)(k0 + bk) * NH_ + colBase + bnp);
      const uint4* srcZ = (const uint4*)(Whz + (long)(k0 + bk) * NH_ + colBase + bnp);
      uint4 r0 = srcR[0], r1 = srcR[1], z0 = srcZ[0], z1 = srcZ[1];
      const bf16_t* er0 = (const bf16_t*)&r0; const bf16_t* er1 = (const bf16_t*)&r1;
      const bf16_t* ez0 = (const bf16_t*)&z0; const bf16_t* ez1 = (const bf16_t*)&z1;
#pragma unroll
      for (int j = 0; j < 8; ++j) {
        sBr[(bnp + j)     * LDK + bk] = er0[j];
        sBr[(bnp + 8 + j) * LDK + bk] = er1[j];
        sBz[(bnp + j)     * LDK + bk] = ez0[j];
        sBz[(bnp + 8 + j) * LDK + bk] = ez1[j];
      }
    }
    __syncthreads();
    bf16x16 aF = load_frag(&sA[(wr * 16) * LDK], LDK);
#pragma unroll
    for (int j = 0; j < 4; ++j) {
      bf16x16 bR = load_frag(&sBr[(wc * 64 + j * 16) * LDK], LDK);
      bf16x16 bZ = load_frag(&sBz[(wc * 64 + j * 16) * LDK], LDK);
      accR[j] = wmma_bf16(aF, bR, accR[j]);
      accZ[j] = wmma_bf16(aF, bZ, accZ[j]);
    }
    __syncthreads();
  }

  const int mlane = (lane >> 4) * 8;
  const int nlane = lane & 15;
#pragma unroll
  for (int j = 0; j < 4; ++j) {
    int n = colBase + wc * 64 + j * 16 + nlane;
#pragma unroll
    for (int e = 0; e < 8; ++e) {
      int m = wr * 16 + mlane + e;                       // batch index
      long gidx = ((long)m * S_ + t) * NH_ + n;          // (B,S,NH)
      long hidx = (long)m * NH_ + n;                     // (B,NH)
      float r = sigmoidf_(accR[j][e] + (float)xr[gidx]);
      float z = sigmoidf_(accZ[j][e] + (float)xz[gidx]);
      RH[hidx]   = to_bf16(r * Hf[hidx]);
      Zbuf[hidx] = z;
    }
  }
}

// ---------------------------------------------------------------------------
// Step stage 2: h_hat = tanh(xh_t + (r*H)@W_hh);
//               H_new = z*H + (1-z)*h_hat; write to outs + update state.
// Weight tile staged with the Tensor Data Mover (TENSOR_LOAD_TO_LDS),
// fenced with s_wait_tensorcnt; fragments transposed on LDS read.
// ---------------------------------------------------------------------------
__global__ __launch_bounds__(256) void step_h_kernel(
    const bf16_t* __restrict__ RH, const bf16_t* __restrict__ Whh,
    const float* __restrict__ Zbuf, float* __restrict__ Hf,
    bf16_t* __restrict__ Hb, float* __restrict__ out, int t) {
  constexpr int LDK = 40;
  __shared__ __attribute__((aligned(16))) bf16_t sA[64 * LDK];
  __shared__ __attribute__((aligned(16))) bf16_t sB[32 * 128];   // row-major [k][n]

  const int colBase = blockIdx.x * 128;
  const int tid = threadIdx.x;
  const int lane = tid & 31;
  const int wave = tid >> 5;
  const int wr = wave >> 1;
  const int wc = wave & 1;

  f32x8 acc[4];
#pragma unroll
  for (int j = 0; j < 4; ++j) acc[j] = zero8();

  const int arow = tid >> 2;
  const int akp  = (tid & 3) * 8;
  const unsigned sB_lds = (unsigned)(unsigned long long)&sB[0];

  for (int k0 = 0; k0 < NH_; k0 += 32) {
    // ---- stage A tile (manual) ----
    {
      uint4 d = *(const uint4*)(RH + (long)arow * NH_ + k0 + akp);
      const bf16_t* e = (const bf16_t*)&d;
#pragma unroll
      for (int j = 0; j < 8; ++j) sA[arow * LDK + akp + j] = e[j];
    }
    // ---- stage B tile via TDM: 32 rows x 128 cols of W_hh ----
    if (wave == 0) {
      tdm_load_2d_bf16(sB_lds, Whh + (long)k0 * NH_ + colBase,
                       /*tensor_d0=*/NH_, /*tensor_d1=*/NH_,
                       /*d0_stride=*/NH_, /*tile_d0=*/128, /*tile_d1=*/32);
      __builtin_amdgcn_s_wait_tensorcnt(0);
    }
    __syncthreads();

    bf16x16 aF = load_frag(&sA[(wr * 16) * LDK], LDK);
#pragma unroll
    for (int j = 0; j < 4; ++j) {
      bf16x16 bF = load_frag_rowmajor(&sB[wc * 64 + j * 16], 128);
      acc[j] = wmma_bf16(aF, bF, acc[j]);
    }
    __syncthreads();
  }

  const int mlane = (lane >> 4) * 8;
  const int nlane = lane & 15;
#pragma unroll
  for (int j = 0; j < 4; ++j) {
    int n = colBase + wc * 64 + j * 16 + nlane;
#pragma unroll
    for (int e = 0; e < 8; ++e) {
      int m = wr * 16 + mlane + e;
      long gidx = ((long)m * S_ + t) * NH_ + n;
      long hidx = (long)m * NH_ + n;
      float hhat = tanhf(acc[j][e] + out[gidx]);   // out holds xh_t pre-act
      float z = Zbuf[hidx];
      float hn = z * Hf[hidx] + (1.0f - z) * hhat;
      out[gidx] = hn;                               // outs[b, t, :]
      Hf[hidx]  = hn;
      Hb[hidx]  = to_bf16(hn);
    }
  }
}

__global__ void final_copy_kernel(const float* __restrict__ Hf, float* __restrict__ dst) {
  int i = blockIdx.x * blockDim.x + threadIdx.x;
  if (i < B_ * NH_) dst[i] = Hf[i];
}

// ---------------------------------------------------------------------------
extern "C" void kernel_launch(void* const* d_in, const int* in_sizes, int n_in,
                              void* d_out, int out_size, void* d_ws, size_t ws_size,
                              hipStream_t stream) {
  const float* X   = (const float*)d_in[0];
  const float* H0  = (const float*)d_in[1];
  const float* Wxz = (const float*)d_in[2];
  const float* Whz = (const float*)d_in[3];
  const float* bz  = (const float*)d_in[4];
  const float* Wxr = (const float*)d_in[5];
  const float* Whr = (const float*)d_in[6];
  const float* br  = (const float*)d_in[7];
  const float* Wxh = (const float*)d_in[8];
  const float* Whh = (const float*)d_in[9];
  const float* bh  = (const float*)d_in[10];

  char* ws = (char*)d_ws;
  const long WN = (long)NI_ * NH_;                              // 1M elems
  bf16_t* wb = (bf16_t*)ws;                                     // 12 MB
  bf16_t* xz = (bf16_t*)(ws + 12l * 1024 * 1024);               // 64 MB
  bf16_t* xr = (bf16_t*)(ws + 76l * 1024 * 1024);               // 64 MB
  float*  Hf = (float*) (ws + 140l * 1024 * 1024);              // 256 KB
  bf16_t* Hb = (bf16_t*)(ws + 140l * 1024 * 1024 + 256 * 1024); // 128 KB
  bf16_t* RH = (bf16_t*)(ws + 140l * 1024 * 1024 + 384 * 1024); // 128 KB
  float*  Z  = (float*) (ws + 140l * 1024 * 1024 + 512 * 1024); // 256 KB

  float* out = (float*)d_out;   // (B,S,NH) outs, then H_final

  {
    long total = 6 * WN + (long)B_ * NH_;
    unsigned blocks = (unsigned)((total + 255) / 256);
    prep_kernel<<<dim3(blocks), 256, 0, stream>>>(Wxz, Whz, Wxr, Whr, Wxh, Whh,
                                                  H0, wb, Hf, Hb);
  }

  // xz, xr -> ws (bf16); xh -> d_out (fp32, overwritten per step with H_new)
  gemm_x_kernel<<<dim3(B_ * S_ / 128, NH_ / 128, 3), 256, 0, stream>>>(
      X, wb, bz, br, bh, xz, xr, out);

  const bf16_t* Whz_b = wb + 3 * WN;
  const bf16_t* Whr_b = wb + 4 * WN;
  const bf16_t* Whh_b = wb + 5 * WN;
  for (int t = 0; t < S_; ++t) {
    step_rz_kernel<<<dim3(NH_ / 128), 256, 0, stream>>>(Hb, Hf, Whr_b, Whz_b,
                                                        xr, xz, RH, Z, t);
    step_h_kernel<<<dim3(NH_ / 128), 256, 0, stream>>>(RH, Whh_b, Z, Hf, Hb,
                                                       out, t);
  }
  final_copy_kernel<<<dim3((B_ * NH_ + 255) / 256), 256, 0, stream>>>(
      Hf, out + (long)B_ * S_ * NH_);
}
